// BigramLanguageModel_81604378624169
// MI455X (gfx1250) — matrix operand order repaired and checked by reading, not compiled, not run on hardware
//
#include <hip/hip_runtime.h>
#include <hip/hip_bf16.h>
#include <math.h>

#define VOCAB 65
#define VPAD  80
#define TSEQ  256
#define DEMB  384
#define NH    6
#define HSZ   64
#define NL    6
#define BB    64
#define BT    (BB*TSEQ)    // 16384 rows
#define DFF   (4*DEMB)     // 1536

typedef unsigned short u16;
typedef unsigned int   u32;
typedef __attribute__((ext_vector_type(16))) __bf16 v16bf;
typedef __attribute__((ext_vector_type(8)))  float  v8f;

union FragU { u32 u[8]; v16bf v; };
union AccU  { v8f  v; float f[8]; };

__device__ __forceinline__ u16 f2bf(float f){
    u32 u; __builtin_memcpy(&u, &f, 4);
    u32 r = u + 0x7FFFu + ((u >> 16) & 1u);   // round-to-nearest-even
    return (u16)(r >> 16);
}

// ---------------------------------------------------------------------------
// WMMA fragment loaders (gfx1250 wave32; contiguous k order k = 16*(lane/16) + s for A and B alike)
// All loaders below read K-contiguous memory -> coalescible 32/128-bit loads.
// ---------------------------------------------------------------------------

// A fragment 16x32 bf16 from row-major memory (lda elements, K contiguous).
// lanes 0-15: M=lane, K=0..15; lanes 16-31: M=lane-16, K=16..31
__device__ __forceinline__ v16bf load_a_frag(const u16* A, int lda, int lane){
    FragU fr;
    int m  = lane & 15;
    int kh = (lane < 16) ? 0 : 16;
    const u16* rowp = A + (size_t)m * lda;
#pragma unroll
    for (int i = 0; i < 8; ++i){
        int k = kh + i * 2;
        u32 val; __builtin_memcpy(&val, rowp + k, 4);
        fr.u[i] = val;
    }
    return fr.v;
}

// B fragment 32x16 bf16 where memory holds B^T row-major: element (k,n) at
// Bt[n*ld + k] (K contiguous per column n). lanes 0-15: N=lane, K=0..15;
// lanes 16-31: N=lane-16, K=16..31.
__device__ __forceinline__ v16bf load_b_frag_kt(const u16* Bt, int ld, int lane){
    FragU fr;
    int n  = lane & 15;
    int kh = (lane < 16) ? 0 : 16;
    const u16* rowp = Bt + (size_t)n * ld + kh;
#pragma unroll
    for (int i = 0; i < 8; ++i){
        u32 val; __builtin_memcpy(&val, rowp + 2 * i, 4);
        fr.u[i] = val;
    }
    return fr.v;
}

// ---------------------------------------------------------------------------
// Wide WMMA GEMM: each wave computes a 16 x (16*TN) strip of C.
//   A  : [M,K]   bf16 row-major
//   Bt : [Npad,K] bf16 (pre-transposed weights, K contiguous)
// OUTMODE: 0 = bf16 store, 1 = f32 store, 2 = f32 residual add,
//          3 = bf16 store transposed to [B,H,HS,T] (for the V projection)
// ---------------------------------------------------------------------------
template<int TN, int OUTMODE, bool BIAS, bool RELU>
__global__ void gemm_tile(const u16* __restrict__ A, const u16* __restrict__ Bt,
                          const float* __restrict__ bias,
                          float* __restrict__ Cf, u16* __restrict__ Cb,
                          int N, int K){
    int lane = threadIdx.x;
    int xt = blockIdx.x, yt = blockIdx.y;
    const u16* Ap = A + (size_t)yt * 16 * K;
    AccU acc[TN];
#pragma unroll
    for (int j = 0; j < TN; ++j)
#pragma unroll
        for (int i = 0; i < 8; ++i) acc[j].f[i] = 0.0f;

    for (int kb = 0; kb < K; kb += 32){
        __builtin_prefetch(Ap + kb + 64, 0, 1);   // global_prefetch_b8
        v16bf a = load_a_frag(Ap + kb, K, lane);
#pragma unroll
        for (int j = 0; j < TN; ++j){
            const u16* Bp = Bt + (size_t)(xt * TN + j) * 16 * K + kb;
            v16bf b = load_b_frag_kt(Bp, K, lane);
            acc[j].v = __builtin_amdgcn_wmma_f32_16x16x32_bf16(
                false, a, false, b, (short)0, acc[j].v, false, false);
        }
    }

    int rbase = yt * 16 + ((lane < 16) ? 0 : 8);
#pragma unroll
    for (int j = 0; j < TN; ++j){
        int col = (xt * TN + j) * 16 + (lane & 15);
        if (col >= N) continue;
#pragma unroll
        for (int i = 0; i < 8; ++i){
            float vv = acc[j].f[i];
            if (BIAS) vv += bias[col];
            if (RELU) vv = fmaxf(vv, 0.0f);
            int row = rbase + i;
            if (OUTMODE == 3){
                int t = row & (TSEQ - 1), bb = row >> 8;
                int h = col / HSZ, hs = col % HSZ;
                Cb[((size_t)((bb * NH + h) * HSZ + hs)) * TSEQ + t] = f2bf(vv);
            } else {
                size_t o = (size_t)row * N + col;
                if (OUTMODE == 0)      Cb[o] = f2bf(vv);
                else if (OUTMODE == 1) Cf[o] = vv;
                else                   Cf[o] += vv;
            }
        }
    }
}

// ---------------------------------------------------------------------------
// Fused causal attention: one wave per (16-query tile, head, batch).
// scores = (Q Kt)*scale -> causal softmax -> att @ V, all via WMMA bf16.
// V is pre-transposed to [B,H,HS,T] so both matmuls load K-contiguous.
// ---------------------------------------------------------------------------
__global__ void attn_kernel(const u16* __restrict__ q, const u16* __restrict__ kbuf,
                            const u16* __restrict__ vT, u16* __restrict__ obuf){
    __shared__ float sc[16][TSEQ];       // f32 scores
    __shared__ u16   attb[16][TSEQ];     // bf16 normalized attention
    int lane = threadIdx.x;
    int qt = blockIdx.x, hh = blockIdx.y, b = blockIdx.z;

    // zero the bf16 attention tile (masked / padded columns must be 0)
    u32* az = (u32*)&attb[0][0];
    for (int j = lane; j < (16 * TSEQ) / 2; j += 32) az[j] = 0u;

    size_t rowq = (size_t)b * TSEQ + qt * 16;
    int col0 = hh * HSZ;
    const u16* Aq = q + rowq * DEMB + col0;
    v16bf a0 = load_a_frag(Aq,      DEMB, lane);   // K = 0..31 of head dim
    v16bf a1 = load_a_frag(Aq + 32, DEMB, lane);   // K = 32..63
    const float scale = 0.05103103630798287f;      // 1/sqrt(384) (ref uses D!)

    for (int jt = 0; jt <= qt; ++jt){
        const u16* Bk = kbuf + ((size_t)b * TSEQ + jt * 16) * DEMB + col0;
        v16bf b0 = load_b_frag_kt(Bk,      DEMB, lane);
        v16bf b1 = load_b_frag_kt(Bk + 32, DEMB, lane);
        AccU s;
#pragma unroll
        for (int i = 0; i < 8; ++i) s.f[i] = 0.0f;
        s.v = __builtin_amdgcn_wmma_f32_16x16x32_bf16(false, a0, false, b0,
                                                      (short)0, s.v, false, false);
        s.v = __builtin_amdgcn_wmma_f32_16x16x32_bf16(false, a1, false, b1,
                                                      (short)0, s.v, false, false);
        int c = lane & 15;
        int rb = (lane < 16) ? 0 : 8;
#pragma unroll
        for (int i = 0; i < 8; ++i) sc[rb + i][jt * 16 + c] = s.f[i] * scale;
    }
    __syncthreads();

    // causal softmax: 2 lanes per row, 128 columns each; never read cols > row
    {
        int row   = lane & 15;
        int half  = lane >> 4;
        int limit = qt * 16 + row + 1;            // #valid key columns
        int cs = half * 128;
        int ce = cs + 128; if (ce > limit) ce = limit;
        float mx = -3.4e38f;
        for (int c = cs; c < ce; ++c) mx = fmaxf(mx, sc[row][c]);
        mx = fmaxf(mx, __shfl_xor(mx, 16, 32));
        float sum = 0.0f;
        for (int c = cs; c < ce; ++c) sum += __expf(sc[row][c] - mx);
        sum += __shfl_xor(sum, 16, 32);
        float inv = 1.0f / sum;
        for (int c = cs; c < ce; ++c) attb[row][c] = f2bf(__expf(sc[row][c] - mx) * inv);
    }
    __syncthreads();

    // out[16, 64] = att[16, keys] @ V[keys, 64]; padded att cols are zero.
    // vT is [B,H,HS,T]: B-fragment columns n are HS channels, K (tokens) contiguous.
    int ksteps = (qt * 16 + 16 + 31) >> 5;
    const u16* Vh = vT + ((size_t)(b * NH + hh) * HSZ) * TSEQ;
    u16* Ob = obuf + rowq * DEMB + col0;
#pragma unroll
    for (int nt = 0; nt < 4; ++nt){
        AccU o;
#pragma unroll
        for (int i = 0; i < 8; ++i) o.f[i] = 0.0f;
        for (int ks = 0; ks < ksteps; ++ks){
            v16bf aa = load_a_frag(&attb[0][ks * 32], TSEQ, lane);
            v16bf bb = load_b_frag_kt(Vh + (size_t)nt * 16 * TSEQ + ks * 32, TSEQ, lane);
            o.v = __builtin_amdgcn_wmma_f32_16x16x32_bf16(false, aa, false, bb,
                                                          (short)0, o.v, false, false);
        }
        int c = lane & 15;
        int rb = (lane < 16) ? 0 : 8;
#pragma unroll
        for (int i = 0; i < 8; ++i)
            Ob[(size_t)(rb + i) * DEMB + nt * 16 + c] = f2bf(o.f[i]);
    }
}

// ---------------------------------------------------------------------------
// LayerNorm: one 128-thread block per row, writes bf16
// ---------------------------------------------------------------------------
__global__ void ln_kernel(const float* __restrict__ x, const float* __restrict__ g,
                          const float* __restrict__ bet, u16* __restrict__ out){
    int row = blockIdx.x;
    int tid = threadIdx.x;
    const float* xr = x + (size_t)row * DEMB;
    float s = 0.0f, s2 = 0.0f;
    for (int d = tid; d < DEMB; d += 128){ float v = xr[d]; s += v; s2 += v * v; }
#pragma unroll
    for (int off = 16; off; off >>= 1){
        s  += __shfl_xor(s,  off, 32);
        s2 += __shfl_xor(s2, off, 32);
    }
    __shared__ float as[4], bs[4];
    int w = tid >> 5;
    if ((tid & 31) == 0){ as[w] = s; bs[w] = s2; }
    __syncthreads();
    s  = as[0] + as[1] + as[2] + as[3];
    s2 = bs[0] + bs[1] + bs[2] + bs[3];
    float mean = s * (1.0f / DEMB);
    float var  = s2 * (1.0f / DEMB) - mean * mean;
    float rstd = rsqrtf(var + 1e-5f);
    u16* orow = out + (size_t)row * DEMB;
    for (int d = tid; d < DEMB; d += 128)
        orow[d] = f2bf((xr[d] - mean) * rstd * g[d] + bet[d]);
}

__global__ void embed_kernel(const int* __restrict__ idx, const float* __restrict__ tok,
                             const float* __restrict__ pos, float* __restrict__ x){
    int i = blockIdx.x * blockDim.x + threadIdx.x;
    if (i >= BT * DEMB) return;
    int d = i % DEMB; int bt = i / DEMB; int t = bt % TSEQ;
    x[i] = tok[(size_t)idx[bt] * DEMB + d] + pos[(size_t)t * DEMB + d];
}

// fp32 [L,K,N] -> bf16 transposed [L,Npad,K], zero-padded columns
__global__ void cvt_t_kernel(const float* __restrict__ src, u16* __restrict__ dst,
                             int K, int N, int Npad, int total){
    int i = blockIdx.x * blockDim.x + threadIdx.x;
    if (i >= total) return;
    int k = i % K;
    int n = (i / K) % Npad;
    int l = i / (K * Npad);
    float v = (n < N) ? src[(size_t)l * K * N + (size_t)k * N + n] : 0.0f;
    dst[i] = f2bf(v);
}

__global__ void loss_row_kernel(const float* __restrict__ logits,
                                const int* __restrict__ tgt, float* __restrict__ rl){
    int r = blockIdx.x * blockDim.x + threadIdx.x;
    if (r >= BT) return;
    const float* L = logits + (size_t)r * VOCAB;
    float mx = -3.4e38f;
    for (int j = 0; j < VOCAB; ++j) mx = fmaxf(mx, L[j]);
    float s = 0.0f;
    for (int j = 0; j < VOCAB; ++j) s += __expf(L[j] - mx);
    rl[r] = (logf(s) + mx) - L[tgt[r]];
}

__global__ void loss_reduce_kernel(const float* __restrict__ rl, float* __restrict__ out){
    __shared__ float sm[256];
    int tid = threadIdx.x;
    float s = 0.0f;
    for (int j = tid; j < BT; j += 256) s += rl[j];
    sm[tid] = s; __syncthreads();
    for (int off = 128; off; off >>= 1){
        if (tid < off) sm[tid] += sm[tid + off];
        __syncthreads();
    }
    if (tid == 0) out[0] = sm[0] * (1.0f / BT);
}

// ---------------------------------------------------------------------------
extern "C" void kernel_launch(void* const* d_in, const int* in_sizes, int n_in,
                              void* d_out, int out_size, void* d_ws, size_t ws_size,
                              hipStream_t stream){
    (void)in_sizes; (void)n_in; (void)out_size; (void)ws_size;
    const int*   idx   = (const int*)  d_in[0];
    const int*   tgt   = (const int*)  d_in[1];
    const float* tok   = (const float*)d_in[2];
    const float* pos   = (const float*)d_in[3];
    const float* ln1g  = (const float*)d_in[4];
    const float* ln1b  = (const float*)d_in[5];
    const float* wq    = (const float*)d_in[6];
    const float* wk    = (const float*)d_in[7];
    const float* wv    = (const float*)d_in[8];
    const float* wo    = (const float*)d_in[9];
    const float* bo    = (const float*)d_in[10];
    const float* ln2g  = (const float*)d_in[11];
    const float* ln2b  = (const float*)d_in[12];
    const float* w1    = (const float*)d_in[13];
    const float* b1    = (const float*)d_in[14];
    const float* w2    = (const float*)d_in[15];
    const float* b2    = (const float*)d_in[16];
    const float* lnfg  = (const float*)d_in[17];
    const float* lnfb  = (const float*)d_in[18];
    const float* hw    = (const float*)d_in[19];
    const float* hb_f  = (const float*)d_in[20];

    char* ws = (char*)d_ws;
    size_t off = 0;
    auto alloc = [&](size_t bytes) -> char* {
        char* p = ws + off;
        off += (bytes + 255) & ~(size_t)255;
        return p;
    };
    float* x   = (float*)alloc((size_t)BT * DEMB * 4);
    u16*   hbb = (u16*)  alloc((size_t)BT * DEMB * 2);
    char*  un  =         alloc((size_t)BT * DFF  * 2);   // qkv OR mlp-mid
    u16*   qb  = (u16*)un;
    u16*   kb  = qb + (size_t)BT * DEMB;
    u16*   vT  = kb + (size_t)BT * DEMB;                 // [B,H,HS,T]
    u16*   mid = (u16*)un;
    u16*   ob  = (u16*)  alloc((size_t)BT * DEMB * 2);
    u16*   wqb = (u16*)  alloc((size_t)NL * DEMB * DEMB * 2);   // transposed
    u16*   wkb = (u16*)  alloc((size_t)NL * DEMB * DEMB * 2);
    u16*   wvb = (u16*)  alloc((size_t)NL * DEMB * DEMB * 2);
    u16*   wob = (u16*)  alloc((size_t)NL * DEMB * DEMB * 2);
    u16*   w1b = (u16*)  alloc((size_t)NL * DFF  * DEMB * 2);   // [L,1536,384]
    u16*   w2b = (u16*)  alloc((size_t)NL * DEMB * DFF  * 2);   // [L,384,1536]
    u16*   hwb = (u16*)  alloc((size_t)VPAD * DEMB * 2);        // [80,384] padded
    float* rl  = (float*)alloc((size_t)BT * 4);

    auto cvtT = [&](const float* s, u16* d, int K, int N, int Npad, int layers){
        int total = layers * Npad * K;
        cvt_t_kernel<<<(total + 255) / 256, 256, 0, stream>>>(s, d, K, N, Npad, total);
    };
    cvtT(wq, wqb, DEMB, DEMB, DEMB, NL);
    cvtT(wk, wkb, DEMB, DEMB, DEMB, NL);
    cvtT(wv, wvb, DEMB, DEMB, DEMB, NL);
    cvtT(wo, wob, DEMB, DEMB, DEMB, NL);
    cvtT(w1, w1b, DEMB, DFF,  DFF,  NL);
    cvtT(w2, w2b, DFF,  DEMB, DEMB, NL);
    cvtT(hw, hwb, DEMB, VOCAB, VPAD, 1);

    embed_kernel<<<(BT * DEMB + 255) / 256, 256, 0, stream>>>(idx, tok, pos, x);

    dim3 gD(DEMB / 64, BT / 16);     // N=384, 4 tiles/wave
    dim3 gF(DFF  / 64, BT / 16);     // N=1536, 4 tiles/wave
    for (int l = 0; l < NL; ++l){
        size_t wOff  = (size_t)l * DEMB * DEMB;
        size_t wfOff = (size_t)l * DEMB * DFF;

        ln_kernel<<<BT, 128, 0, stream>>>(x, ln1g + l * DEMB, ln1b + l * DEMB, hbb);
        gemm_tile<4, 0, false, false><<<gD, 32, 0, stream>>>(hbb, wqb + wOff, nullptr, nullptr, qb, DEMB, DEMB);
        gemm_tile<4, 0, false, false><<<gD, 32, 0, stream>>>(hbb, wkb + wOff, nullptr, nullptr, kb, DEMB, DEMB);
        gemm_tile<4, 3, false, false><<<gD, 32, 0, stream>>>(hbb, wvb + wOff, nullptr, nullptr, vT, DEMB, DEMB);

        attn_kernel<<<dim3(TSEQ / 16, NH, BB), 32, 0, stream>>>(qb, kb, vT, ob);

        gemm_tile<4, 2, true, false><<<gD, 32, 0, stream>>>(ob, wob + wOff, bo + l * DEMB, x, nullptr, DEMB, DEMB);

        ln_kernel<<<BT, 128, 0, stream>>>(x, ln2g + l * DEMB, ln2b + l * DEMB, hbb);
        gemm_tile<4, 0, true, true ><<<gF, 32, 0, stream>>>(hbb, w1b + wfOff, b1 + l * DFF, nullptr, mid, DFF, DEMB);
        gemm_tile<4, 2, true, false><<<gD, 32, 0, stream>>>(mid, w2b + wfOff, b2 + l * DEMB, x, nullptr, DEMB, DFF);
    }

    ln_kernel<<<BT, 128, 0, stream>>>(x, lnfg, lnfb, hbb);
    float* logits = (float*)d_out;
    gemm_tile<1, 1, true, false><<<dim3(VPAD / 16, BT / 16), 32, 0, stream>>>(
        hbb, hwb, hb_f, logits, nullptr, VOCAB, DEMB);

    loss_row_kernel<<<(BT + 255) / 256, 256, 0, stream>>>(logits, tgt, rl);
    loss_reduce_kernel<<<1, 256, 0, stream>>>(rl, logits + (size_t)BT * VOCAB);
}
